// LocalCtxAttRanker_12910671692200
// MI455X (gfx1250) — compile-verified, matrix operand order checked by reading, not run on hardware
//
#include <hip/hip_runtime.h>
#include <hip/hip_bf16.h>
#include <math.h>

typedef __attribute__((ext_vector_type(16))) __bf16 v16bf;
typedef __attribute__((ext_vector_type(8)))  __bf16 v8bf;
typedef __attribute__((ext_vector_type(4)))  __bf16 v4bf;
typedef __attribute__((ext_vector_type(8)))  float  v8f;
typedef __attribute__((ext_vector_type(4)))  float  v4f;

#define BATCH   128
#define N_ENT   256
#define N_WORDS 1024
#define DIM     300
#define DPAD    320      // K padded to multiple of 32 (bf16 WMMA K)
#define DSTRIDE 328      // LDS row stride (elems): 16B-aligned frags, conflict-free
#define TILE_W  128      // words per workgroup
#define ETILE   16       // entities per WMMA tile
#define NT      (N_ENT / ETILE)
#define KSTEPS  (DPAD / 32)
#define TOPK    15

// Pack a 16x32 bf16 A/B WMMA fragment from LDS per the CDNA5 layout:
// lane L: row/col = L&15 ; K-base = (L<16 ? 0 : 8);
// v[0..7] hold K = base..base+7, v[8..15] hold K = base+16..base+23.
__device__ __forceinline__ v16bf pack_frag(const __bf16* p) {
    v8bf lo = *(const v8bf*)(p);        // ds_load_b128
    v8bf hi = *(const v8bf*)(p + 16);   // ds_load_b128
    v16bf r;
#pragma unroll
    for (int i = 0; i < 8; ++i) { r[i] = lo[i]; r[i + 8] = hi[i]; }
    return r;
}

// ---------------------------------------------------------------------------
// Pass 1: fused  S = (E * att_diag) @ C^T  with max-over-entities epilogue.
// Grid: (BATCH, N_WORDS/TILE_W), 256 threads (8 waves).
// Ping-pong entity tiles: global->regs before WMMA block, regs->LDS after.
// ---------------------------------------------------------------------------
__global__ __launch_bounds__(256)
void ranker_scores_wmma(const float* __restrict__ ent,
                        const float* __restrict__ ctx,
                        const float* __restrict__ att_diag,
                        float* __restrict__ tok_scores) {
    __shared__ __attribute__((aligned(16))) __bf16 sCtx[TILE_W * DSTRIDE];
    __shared__ __attribute__((aligned(16))) __bf16 sEnt[2][ETILE * DSTRIDE];

    const int b    = blockIdx.x;
    const int wt   = blockIdx.y;
    const int tid  = threadIdx.x;
    const int lane = tid & 31;
    const int wave = tid >> 5;

    const float* ctxB = ctx + ((size_t)b * N_WORDS + (size_t)wt * TILE_W) * DIM;
    const float* entB = ent + (size_t)b * N_ENT * DIM;

    // Stage context tile as bf16 into LDS (zero-pad K to DPAD). One-time cost.
    for (int idx = tid; idx < TILE_W * DPAD; idx += 256) {
        int r = idx / DPAD, c = idx - r * DPAD;
        float v = (c < DIM) ? ctxB[r * DIM + c] : 0.0f;
        sCtx[r * DSTRIDE + c] = (__bf16)v;
    }

    // Fixed per-thread staging slice for entity tiles: row er, 20 contiguous cols.
    const int  er    = tid >> 4;            // 0..15
    const int  ec0   = (tid & 15) * 20;     // 0,20,...,300 (16B-aligned in f32)
    const bool evld  = (ec0 < DIM);         // ec0==300 slice is pure zero-pad
    v4f ad4[5];
#pragma unroll
    for (int j = 0; j < 5; ++j)
        ad4[j] = evld ? *(const v4f*)(att_diag + ec0 + 4 * j) : (v4f){0, 0, 0, 0};

    // Prefill entity tile 0 into sEnt[0].
    {
        const float* eB = entB + (size_t)er * DIM + ec0;
        __bf16* dst = &sEnt[0][er * DSTRIDE + ec0];
#pragma unroll
        for (int j = 0; j < 5; ++j) {
            v4f w = evld ? (*(const v4f*)(eB + 4 * j)) * ad4[j] : (v4f){0, 0, 0, 0};
            v4bf p; p[0] = (__bf16)w[0]; p[1] = (__bf16)w[1];
                    p[2] = (__bf16)w[2]; p[3] = (__bf16)w[3];
            *(v4bf*)(dst + 4 * j) = p;
        }
    }
    __syncthreads();

    // Pre-load all 10 B fragments for this wave's 16-word subtile.
    const int colword = wave * 16 + (lane & 15);
    const int kbase   = (lane < 16) ? 0 : 8;
    v16bf bfrag[KSTEPS];
#pragma unroll
    for (int kk = 0; kk < KSTEPS; ++kk)
        bfrag[kk] = pack_frag(&sCtx[colword * DSTRIDE + kk * 32 + kbase]);

    const int arow = lane & 15;
    float runmax = -INFINITY;

    for (int et = 0; et < NT; ++et) {
        const int  cur  = et & 1;
        const bool more = (et + 1 < NT);

        // Issue next tile's global loads now; consume after the WMMA block.
        v4f stage[5];
        if (more) {
            const float* eB = entB + (size_t)(et + 1) * ETILE * DIM
                                   + (size_t)er * DIM + ec0;
#pragma unroll
            for (int j = 0; j < 5; ++j)
                stage[j] = evld ? *(const v4f*)(eB + 4 * j) : (v4f){0, 0, 0, 0};
        }

        // 10 WMMAs over K, A-fragment double-buffered, two accumulator chains.
        const __bf16* aBase = &sEnt[cur][arow * DSTRIDE + kbase];
        v16bf a = pack_frag(aBase);
        v8f acc0 = {}, acc1 = {};
#pragma unroll
        for (int kk = 0; kk < KSTEPS; ++kk) {
            v16bf an;
            if (kk + 1 < KSTEPS) an = pack_frag(aBase + (kk + 1) * 32);
            if (kk & 1)
                acc1 = __builtin_amdgcn_wmma_f32_16x16x32_bf16(
                    false, a, false, bfrag[kk], (short)0, acc1, false, false);
            else
                acc0 = __builtin_amdgcn_wmma_f32_16x16x32_bf16(
                    false, a, false, bfrag[kk], (short)0, acc0, false, false);
            a = an;
        }

        // Per-word (column) max over this 16-entity tile.
        float m = acc0[0] + acc1[0];
#pragma unroll
        for (int i = 1; i < 8; ++i) m = fmaxf(m, acc0[i] + acc1[i]);
        m = fmaxf(m, __shfl_xor(m, 16, 32));
        runmax = fmaxf(runmax, m);

        // Convert staged tile and store into the other buffer.
        if (more) {
            __bf16* dst = &sEnt[1 - cur][er * DSTRIDE + ec0];
#pragma unroll
            for (int j = 0; j < 5; ++j) {
                v4f w = stage[j] * ad4[j];
                v4bf p; p[0] = (__bf16)w[0]; p[1] = (__bf16)w[1];
                        p[2] = (__bf16)w[2]; p[3] = (__bf16)w[3];
                *(v4bf*)(dst + 4 * j) = p;
            }
        }
        __syncthreads();
    }

    if (lane < 16)
        tok_scores[(size_t)b * N_WORDS + wt * TILE_W + wave * 16 + lane] = runmax;
}

// ---------------------------------------------------------------------------
// Pass 2: per-batch top-15, softmax, weighted ctx vector, final entity dots.
// Grid: BATCH workgroups of 256 threads.
// ---------------------------------------------------------------------------
__global__ __launch_bounds__(256)
void ranker_topk_score(const float* __restrict__ ent,
                       const float* __restrict__ ctx,
                       const float* __restrict__ score_diag,
                       const float* __restrict__ tok_scores,
                       float* __restrict__ out) {
    __shared__ float s[N_WORDS];
    __shared__ float rv[256];
    __shared__ int   ri[256];
    __shared__ float topv[TOPK];
    __shared__ int   topi[TOPK];
    __shared__ float prob[TOPK];
    __shared__ float sctx[DIM];

    const int b = blockIdx.x, tid = threadIdx.x;

    for (int i = tid; i < N_WORDS; i += 256) s[i] = tok_scores[(size_t)b * N_WORDS + i];
    __syncthreads();

    for (int it = 0; it < TOPK; ++it) {
        float best = -INFINITY; int bi = 0;
        for (int i = tid; i < N_WORDS; i += 256)
            if (s[i] > best) { best = s[i]; bi = i; }
        rv[tid] = best; ri[tid] = bi;
        __syncthreads();
        for (int off = 128; off > 0; off >>= 1) {
            if (tid < off && rv[tid + off] > rv[tid]) {
                rv[tid] = rv[tid + off]; ri[tid] = ri[tid + off];
            }
            __syncthreads();
        }
        if (tid == 0) { topv[it] = rv[0]; topi[it] = ri[0]; s[ri[0]] = -INFINITY; }
        __syncthreads();
    }

    if (tid == 0) {  // softmax over 15 (reference's re-normalization is identity)
        float mx = topv[0], sum = 0.0f, e[TOPK];
        for (int j = 0; j < TOPK; ++j) { e[j] = __expf(topv[j] - mx); sum += e[j]; }
        float inv = 1.0f / sum;
        for (int j = 0; j < TOPK; ++j) prob[j] = e[j] * inv;
    }
    __syncthreads();

    const float* ctxB = ctx + (size_t)b * N_WORDS * DIM;
    for (int d = tid; d < DIM; d += 256) {
        float acc = 0.0f;
        for (int j = 0; j < TOPK; ++j) acc += prob[j] * ctxB[(size_t)topi[j] * DIM + d];
        sctx[d] = acc * score_diag[d];
    }
    __syncthreads();

    const float* entB = ent + (size_t)b * N_ENT * DIM;
    const int lane = tid & 31, wave = tid >> 5;
    for (int e = wave; e < N_ENT; e += 8) {
        float acc = 0.0f;
        for (int d = lane; d < DIM; d += 32) acc += entB[(size_t)e * DIM + d] * sctx[d];
#pragma unroll
        for (int off = 16; off > 0; off >>= 1) acc += __shfl_xor(acc, off, 32);
        if (lane == 0) out[(size_t)b * N_ENT + e] = acc;
    }
}

// ---------------------------------------------------------------------------
extern "C" void kernel_launch(void* const* d_in, const int* in_sizes, int n_in,
                              void* d_out, int out_size, void* d_ws, size_t ws_size,
                              hipStream_t stream) {
    const float* ent      = (const float*)d_in[0];  // [128,256,300]
    const float* ctx      = (const float*)d_in[1];  // [128,1024,300]
    const float* att_diag = (const float*)d_in[2];  // [300]
    const float* sc_diag  = (const float*)d_in[3];  // [300]
    // d_in[4] = tok_top_n (15, compile-time constant here)

    float* tok_scores = (float*)d_ws;               // 128*1024 f32 = 512 KB

    dim3 g1(BATCH, N_WORDS / TILE_W);
    ranker_scores_wmma<<<g1, 256, 0, stream>>>(ent, ctx, att_diag, tok_scores);
    ranker_topk_score<<<BATCH, 256, 0, stream>>>(ent, ctx, sc_diag, tok_scores,
                                                 (float*)d_out);
}
